// L4maAttention_13022340841979
// MI455X (gfx1250) — matrix operand order repaired and checked by reading, not compiled, not run on hardware
//
#include <hip/hip_runtime.h>

// ---------------------------------------------------------------------------
// CDNA5 (gfx1250) attention layer: QKV proj + RoPE + paged-KV flash attention
// + O proj.
//   * matmuls: v_wmma_f32_16x16x32_bf16 (wave32)
//   * attention K/V chunks: global_load_async_to_lds_b128 (ASYNCcnt, 2-buffer)
//   * GEMM weight slabs: tensor_load_to_lds (TDM, TENSORcnt, 2-buffer)
// ---------------------------------------------------------------------------

typedef __bf16 bf16_t;
typedef __attribute__((ext_vector_type(16))) __bf16 v16bf;
typedef __attribute__((ext_vector_type(8)))  __bf16 v8bf;
typedef __attribute__((ext_vector_type(8)))  float  v8f;
typedef __attribute__((ext_vector_type(4)))  unsigned u32x4;
typedef __attribute__((ext_vector_type(8)))  unsigned u32x8;

#define BSZ   2
#define QLEN  1024
#define HID   4096
#define NH    32
#define NKV   8
#define HD    128
#define SCTX  4096   // 256 blocks * 16
#define NBLK  256
#define NEWB  64

// ---- bf16 helpers (manual RNE, bit-exact) ---------------------------------
static __device__ __forceinline__ bf16_t f2bf(float x) {
  unsigned u = __builtin_bit_cast(unsigned, x);
  u += 0x7FFFu + ((u >> 16) & 1u);
  unsigned short h = (unsigned short)(u >> 16);
  return __builtin_bit_cast(bf16_t, h);
}
static __device__ __forceinline__ float bf2f(bf16_t b) {
  unsigned short s = __builtin_bit_cast(unsigned short, b);
  unsigned u = ((unsigned)s) << 16;
  return __builtin_bit_cast(float, u);
}
static __device__ __forceinline__ v16bf join8(v8bf lo, v8bf hi) {
  v16bf r;
#pragma unroll
  for (int i = 0; i < 8; ++i) { r[i] = lo[i]; r[i + 8] = hi[i]; }
  return r;
}
static __device__ __forceinline__ v16bf cvt8x2(const float* p0, const float* p1) {
  v8f a = *(const v8f*)p0;
  v8f b = *(const v8f*)p1;
  v16bf r;
#pragma unroll
  for (int i = 0; i < 8; ++i) { r[i] = f2bf(a[i]); r[i + 8] = f2bf(b[i]); }
  return r;
}
static __device__ __forceinline__ v16bf cvt16(const float* p) {
  return cvt8x2(p, p + 8);
}
static __device__ __forceinline__ v8f wmma_bf16(v16bf a, v16bf b, v8f c) {
  // (neg_a, A, neg_b, B, c_mod, C, reuse_a, reuse_b)
  return __builtin_amdgcn_wmma_f32_16x16x32_bf16(false, a, false, b, (short)0, c,
                                                 false, false);
}

// ---- CDNA5 async global -> LDS (ASYNCcnt) ---------------------------------
// LDS operand is the raw LDS byte offset (hardware maps generic LDS addresses
// by truncating to addr[31:0]).
static __device__ __forceinline__ void async_cp16(void* lds_ptr, const void* gaddr) {
  unsigned lds_off = (unsigned)(unsigned long long)lds_ptr;
  asm volatile("global_load_async_to_lds_b128 %0, %1, off"
               :: "v"(lds_off), "v"(gaddr)
               : "memory");
}
static __device__ __forceinline__ void wait_async_le16() {
  asm volatile("s_wait_asynccnt 0x10" ::: "memory");
}
static __device__ __forceinline__ void wait_async_0() {
  asm volatile("s_wait_asynccnt 0x0" ::: "memory");
}

// ---- CDNA5 Tensor Data Mover: 2D tile -> LDS (TENSORcnt) ------------------
static __device__ __forceinline__ void tdm_load_tile(u32x4 g0, u32x8 g1) {
  asm volatile("tensor_load_to_lds %0, %1" :: "s"(g0), "s"(g1) : "memory");
}

// ---------------------------------------------------------------------------
// Generic K-major GEMM:  C[M,N] = A[M,K] * W[N,K]^T
// Workgroup = 256 thr = 8 waves (2 M x 4 N), tile 64(M) x 128(N).
// Weight slabs (128 rows x 64 K fp32) are DMA'd into LDS by the TDM,
// double-buffered; padding (4 dwords per 64-dword row -> stride 68) is done
// by the TDM itself via D#.pad_*.
// ---------------------------------------------------------------------------
#define WLDS_STRIDE 68  // dwords per staged W row (64 data + 4 pad)

template <bool A_BF16, bool OUT_BF16>
__global__ __launch_bounds__(256) void gemm_kn(const void* __restrict__ Av,
                                               const float* __restrict__ W,
                                               void* __restrict__ Cv,
                                               int M, int N, int K) {
  __shared__ float Wlds[2][128 * WLDS_STRIDE];

  const int lane = threadIdx.x & 31;
  const int wave = threadIdx.x >> 5;
  const int ln   = lane & 15;
  const int hi   = lane >> 4;
  const int wm   = wave & 1;
  const int wn   = wave >> 1;
  const int m0   = blockIdx.y * 64 + wm * 32;
  const int nblk = blockIdx.x * 128;
  const int n0   = nblk + wn * 32;

  v8f acc[2][2];
  const v8f vz = {};
#pragma unroll
  for (int i = 0; i < 2; ++i)
#pragma unroll
    for (int j = 0; j < 2; ++j) acc[i][j] = vz;

  const size_t rowA0 = (size_t)(m0 + ln) * (size_t)K;
  const size_t rowA1 = (size_t)(m0 + 16 + ln) * (size_t)K;

  // ---- TDM descriptor group 1 (constant across slabs) ----
  // data_size=2 (4B), pad_enable, pad_interval=5 (64 dwords), pad_amount=3 (4)
  // tensor_dim0=K, tensor_dim1=2^20, tile_dim0=64, tile_dim1=128,
  // tensor_dim0_stride=K
  u32x8 g1;
  g1[0] = (2u << 16) | (1u << 20) | (5u << 22) | (3u << 25);
  g1[1] = ((unsigned)K & 0xFFFFu) << 16;
  g1[2] = ((unsigned)K >> 16) | (0u << 16);          // tensor_dim1 lo16 = 0
  g1[3] = 16u | (64u << 16);                         // tensor_dim1 hi16, tile_dim0
  g1[4] = 128u;                                      // tile_dim1 (tile_dim2=0)
  g1[5] = (unsigned)K;                               // tensor_dim0_stride lo32
  g1[6] = 0u;
  g1[7] = 0u;

  const int nslab = K >> 6;  // 64-K slabs

  auto issue_slab = [&](int slab, int buf) {
    const unsigned long long ga =
        (unsigned long long)(const void*)(W + (size_t)nblk * (size_t)K + (slab << 6));
    const unsigned lds_addr = (unsigned)(unsigned long long)(void*)&Wlds[buf][0];
    u32x4 g0;
    g0[0] = 1u;                       // count=1, user descriptor
    g0[1] = lds_addr;
    g0[2] = (unsigned)ga;
    g0[3] = ((unsigned)(ga >> 32) & 0x01FFFFFFu) | 0x80000000u;  // type=2
    tdm_load_tile(g0, g1);
  };

  if (wave == 0) issue_slab(0, 0);

  for (int s = 0; s < nslab; ++s) {
    const int cur = s & 1;
    if (wave == 0) {
      if (s + 1 < nslab) {
        issue_slab(s + 1, cur ^ 1);
        __builtin_amdgcn_s_wait_tensorcnt((short)1);  // slab s landed (in-order)
      } else {
        __builtin_amdgcn_s_wait_tensorcnt((short)0);
      }
    }
    __syncthreads();  // slab s visible to all waves

    const int k0 = s << 6;
#pragma unroll
    for (int kk = 0; kk < 2; ++kk) {
      const int k = k0 + kk * 32;
      v16bf aF[2], bF[2];
      if (A_BF16) {
        const bf16_t* A  = (const bf16_t*)Av;
        const bf16_t* p0 = A + rowA0 + k + hi * 8;
        const bf16_t* p1 = A + rowA1 + k + hi * 8;
        aF[0] = join8(*(const v8bf*)p0, *(const v8bf*)(p0 + 16));
        aF[1] = join8(*(const v8bf*)p1, *(const v8bf*)(p1 + 16));
      } else {
        const float* A  = (const float*)Av;
        const float* p0 = A + rowA0 + k + hi * 8;
        const float* p1 = A + rowA1 + k + hi * 8;
        __builtin_prefetch((const void*)(p0 + 64), 0, 0);
        aF[0] = cvt8x2(p0, p0 + 16);
        aF[1] = cvt8x2(p1, p1 + 16);
      }
      bF[0] = cvt8x2(&Wlds[cur][(wn * 32 + ln) * WLDS_STRIDE + kk * 32 + hi * 8], &Wlds[cur][(wn * 32 + ln) * WLDS_STRIDE + kk * 32 + hi * 8 + 16]);
      bF[1] = cvt8x2(&Wlds[cur][(wn * 32 + 16 + ln) * WLDS_STRIDE + kk * 32 + hi * 8], &Wlds[cur][(wn * 32 + 16 + ln) * WLDS_STRIDE + kk * 32 + hi * 8 + 16]);
#pragma unroll
      for (int i = 0; i < 2; ++i)
#pragma unroll
        for (int j = 0; j < 2; ++j) acc[i][j] = wmma_bf16(aF[i], bF[j], acc[i][j]);
    }
    __syncthreads();  // all waves done with buf[cur] before it is re-DMA'd
  }

#pragma unroll
  for (int i = 0; i < 2; ++i)
#pragma unroll
    for (int j = 0; j < 2; ++j)
#pragma unroll
      for (int r = 0; r < 8; ++r) {
        const int row = m0 + i * 16 + hi * 8 + r;
        const int col = n0 + j * 16 + ln;
        if (OUT_BF16)
          ((bf16_t*)Cv)[(size_t)row * N + col] = f2bf(acc[i][j][r]);
        else
          ((float*)Cv)[(size_t)row * N + col] = acc[i][j][r];
      }
}

// ---------------------------------------------------------------------------
// RoPE on Q: q_lin (b,s, h*128+d) bf16 -> q_rope (b,h,s,d) bf16
// ---------------------------------------------------------------------------
__global__ __launch_bounds__(256) void rope_q_kernel(const bf16_t* __restrict__ qlin,
                                                     const float* __restrict__ cosv,
                                                     const float* __restrict__ sinv,
                                                     bf16_t* __restrict__ qre) {
  size_t i = (size_t)blockIdx.x * 256 + threadIdx.x;  // (b,s,h,d)
  const int d = (int)(i & 127); i >>= 7;
  const int h = (int)(i & 31);  i >>= 5;
  const int s = (int)(i & 1023);
  const int b = (int)(i >> 10);
  const size_t base = ((size_t)(b * QLEN + s)) * (NH * HD) + h * HD;
  const float x  = bf2f(qlin[base + d]);
  const int   dp = (d < 64) ? d + 64 : d - 64;
  const float xr = bf2f(qlin[base + dp]);
  const float rt = (d < 64) ? -xr : xr;
  const float o  = x * cosv[s * HD + d] + rt * sinv[s * HD + d];
  qre[(((size_t)(b * NH + h)) * QLEN + s) * HD + d] = f2bf(o);
}

// ---------------------------------------------------------------------------
// Build contiguous KV context (without mutating the input cache):
//  kctx  (b, kvh, s, d)  bf16  -- K with RoPE applied to the new blocks
//  vctxT (b, kvh, d, s)  bf16  -- V transposed so P*V B-fragments are K-contig
// ---------------------------------------------------------------------------
__global__ __launch_bounds__(256) void build_kv_kernel(
    const bf16_t* __restrict__ klin, const bf16_t* __restrict__ vlin,
    const float* __restrict__ kst, const float* __restrict__ vst,
    const int* __restrict__ lut, const int* __restrict__ ptrs,
    const float* __restrict__ cosv, const float* __restrict__ sinv,
    bf16_t* __restrict__ kctx, bf16_t* __restrict__ vctxT) {
  int blk = blockIdx.x;           // (b*256 + i)*8 + kvh
  const int kvh = blk & 7;  blk >>= 3;
  const int i   = blk & 255;
  const int b   = blk >> 8;
  const int lb  = lut[b * NBLK + i];
  int newj = -1;
#pragma unroll 8
  for (int j = 0; j < NEWB; ++j)
    if (ptrs[b * NEWB + j] == lb) newj = j;

  for (int e = threadIdx.x; e < 16 * HD; e += 256) {
    const int d    = e & 127;
    const int row  = e >> 7;
    const int sctx = i * 16 + row;
    float kv, vv;
    if (newj >= 0) {
      const int sn = newj * 16 + row;
      const size_t base = ((size_t)(b * QLEN + sn)) * (NKV * HD) + kvh * HD;
      const float x  = bf2f(klin[base + d]);
      const int   dp = (d < 64) ? d + 64 : d - 64;
      const float xr = bf2f(klin[base + dp]);
      const float rt = (d < 64) ? -xr : xr;
      kv = x * cosv[sn * HD + d] + rt * sinv[sn * HD + d];
      vv = bf2f(vlin[base + d]);
    } else {
      const size_t sb = (((size_t)lb * NKV + kvh) * 16 + row) * HD + d;
      kv = kst[sb];
      vv = vst[sb];
    }
    kctx[(((size_t)(b * NKV + kvh)) * SCTX + sctx) * HD + d]  = f2bf(kv);
    vctxT[(((size_t)(b * NKV + kvh)) * HD + d) * SCTX + sctx] = f2bf(vv);
  }
}

// ---------------------------------------------------------------------------
// Flash attention: one WG per (b, head, 64-row q tile). 8 waves = 4(M) x 2(N).
// K/V chunks (128 positions) double-buffered in LDS via async loads; scores
// and P*V on WMMA; online softmax; O accumulated in registers.
// ---------------------------------------------------------------------------
#define PLD 136  // padded LDS row stride (bf16 elems): conflict-free ds reads
#define KLD 136

__global__ __launch_bounds__(256) void attn_kernel(const bf16_t* __restrict__ qre,
                                                   const bf16_t* __restrict__ kctx,
                                                   const bf16_t* __restrict__ vctxT,
                                                   const int* __restrict__ prefix_ptr,
                                                   bf16_t* __restrict__ attn) {
  __shared__ bf16_t Kbuf[2][128 * KLD];
  __shared__ bf16_t Vbuf[2][128 * KLD];
  __shared__ bf16_t Pbuf[64 * PLD];
  __shared__ float  pmx[2][64];
  __shared__ float  psm[2][64];

  const int tid  = threadIdx.x;
  const int lane = tid & 31;
  const int wave = tid >> 5;
  const int ln   = lane & 15;
  const int hi   = lane >> 4;
  const int wm   = wave & 3;   // 0..3 : M split
  const int wn   = wave >> 2;  // 0..1 : N / d-half split

  int bid = blockIdx.x;
  const int qt = bid & 15; bid >>= 4;
  const int h  = bid & 31; bid >>= 5;
  const int b  = bid;
  const int kvh = h >> 2;  // GQA group of 4
  const int s0  = qt * 64;
  const int prefix = prefix_ptr[0];

  const bf16_t* kb = kctx + ((size_t)(b * NKV + kvh)) * SCTX * HD;
  const bf16_t* vb = vctxT + ((size_t)(b * NKV + kvh)) * HD * SCTX;

  // Q fragments for this wave's 16 rows, all 128 d, kept in registers.
  const bf16_t* qb = qre + (((size_t)(b * NH + h)) * QLEN + (s0 + wm * 16 + ln)) * HD;
  v16bf qf[4];
#pragma unroll
  for (int kk = 0; kk < 4; ++kk) {
    const bf16_t* p = qb + kk * 32 + hi * 8;
    qf[kk] = join8(*(const v8bf*)p, *(const v8bf*)(p + 16));
  }

  float m_reg[8], l_reg[8], alpha[8];
  v8f oacc[4];
  const v8f vz = {};
#pragma unroll
  for (int r = 0; r < 8; ++r) { m_reg[r] = -3.0e38f; l_reg[r] = 0.0f; }
#pragma unroll
  for (int t = 0; t < 4; ++t) oacc[t] = vz;

  const int   qrow_base = prefix + s0 + wm * 16;
  const int   kmax      = prefix + s0 + 63;
  const int   nchunks   = (kmax >> 7) + 1;  // chunks of 128 kv positions
  const float scale     = 0.08838834764831845f;  // 1/sqrt(128)

  // stage a 128-position K/V chunk into LDS: 2048 16B units each, 8/thread
  auto stage = [&](int c0, int buf) {
#pragma unroll
    for (int it = 0; it < 8; ++it) {
      const int u   = tid + it * 256;
      const int pos = u >> 4;
      const int d8  = (u & 15) << 3;
      async_cp16(&Kbuf[buf][pos * KLD + d8], kb + (size_t)(c0 + pos) * HD + d8);
    }
#pragma unroll
    for (int it = 0; it < 8; ++it) {
      const int u  = tid + it * 256;
      const int d  = u >> 4;
      const int p8 = (u & 15) << 3;
      async_cp16(&Vbuf[buf][d * KLD + p8], vb + (size_t)d * SCTX + c0 + p8);
    }
  };

  stage(0, 0);

  for (int c = 0; c < nchunks; ++c) {
    const int c0  = c << 7;
    const int cur = c & 1;
    if (c + 1 < nchunks) {
      stage(c0 + 128, cur ^ 1);
      wait_async_le16();  // previous batch (chunk c) fully landed (in-order)
    } else {
      wait_async_0();
    }
    __syncthreads();  // make all waves' async writes to buf[cur] visible

    // ---- scores: S = Q (16x128) x K^T (128 x 64-per-wave), K from LDS ----
    v8f sacc[4];
#pragma unroll
    for (int t = 0; t < 4; ++t) sacc[t] = vz;
#pragma unroll
    for (int kk = 0; kk < 4; ++kk) {
#pragma unroll
      for (int t = 0; t < 4; ++t) {
        const bf16_t* kp = &Kbuf[cur][(wn * 64 + t * 16 + ln) * KLD + kk * 32 + hi * 8];
        sacc[t] = wmma_bf16(qf[kk], join8(*(const v8bf*)kp, *(const v8bf*)(kp + 16)), sacc[t]);
      }
    }

    // ---- mask + scale + per-row max (cross-lane, then cross-wave via LDS) -
    float mx[8];
#pragma unroll
    for (int r = 0; r < 8; ++r) {
      const int qpos = qrow_base + hi * 8 + r;
      float mr = -3.0e38f;
#pragma unroll
      for (int t = 0; t < 4; ++t) {
        const int kvpos = c0 + wn * 64 + t * 16 + ln;
        float sv = sacc[t][r] * scale;
        sv = (kvpos <= qpos) ? sv : -3.0e38f;
        sacc[t][r] = sv;
        mr = fmaxf(mr, sv);
      }
      for (int off = 8; off >= 1; off >>= 1)
        mr = fmaxf(mr, __shfl_xor(mr, off, 32));
      mx[r] = mr;
    }
    if (ln == 0) {
#pragma unroll
      for (int r = 0; r < 8; ++r) pmx[wn][wm * 16 + hi * 8 + r] = mx[r];
    }
    __syncthreads();

    // ---- online softmax: new max, p = exp(s - m), partial sums, P -> LDS --
#pragma unroll
    for (int r = 0; r < 8; ++r) {
      const int row = wm * 16 + hi * 8 + r;
      const float nm = fmaxf(m_reg[r], fmaxf(pmx[0][row], pmx[1][row]));
      alpha[r] = __expf(m_reg[r] - nm);
      m_reg[r] = nm;
      float sum = 0.0f;
#pragma unroll
      for (int t = 0; t < 4; ++t) {
        const float pv = __expf(sacc[t][r] - nm);
        sum += pv;
        Pbuf[row * PLD + wn * 64 + t * 16 + ln] = f2bf(pv);
      }
      for (int off = 8; off >= 1; off >>= 1) sum += __shfl_xor(sum, off, 32);
      if (ln == 0) psm[wn][row] = sum;
    }
#pragma unroll
    for (int t = 0; t < 4; ++t)
#pragma unroll
      for (int r = 0; r < 8; ++r) oacc[t][r] *= alpha[r];
    __syncthreads();

#pragma unroll
    for (int r = 0; r < 8; ++r) {
      const int row = wm * 16 + hi * 8 + r;
      l_reg[r] = l_reg[r] * alpha[r] + psm[0][row] + psm[1][row];
    }

    // ---- O += P (16x128 pos) x V (128 pos x 64-per-wave d), V from LDS ----
#pragma unroll
    for (int kk = 0; kk < 4; ++kk) {
      const bf16_t* pp = Pbuf + (size_t)(wm * 16 + ln) * PLD + kk * 32 + hi * 8;
      const v16bf ap = join8(*(const v8bf*)pp, *(const v8bf*)(pp + 16));
#pragma unroll
      for (int t = 0; t < 4; ++t) {
        const bf16_t* vp = &Vbuf[cur][(wn * 64 + t * 16 + ln) * KLD + kk * 32 + hi * 8];
        oacc[t] = wmma_bf16(ap, join8(*(const v8bf*)vp, *(const v8bf*)(vp + 16)), oacc[t]);
      }
    }
    __syncthreads();  // protect Pbuf/pmx/psm and buf[cur] for reuse
  }

  // ---- normalize and store: attn (b, s, h*128+d) bf16 ---------------------
#pragma unroll
  for (int t = 0; t < 4; ++t)
#pragma unroll
    for (int r = 0; r < 8; ++r) {
      const int srow = s0 + wm * 16 + hi * 8 + r;
      const int col  = h * HD + wn * 64 + t * 16 + ln;
      attn[((size_t)(b * QLEN + srow)) * (NH * HD) + col] =
          f2bf(oacc[t][r] / l_reg[r]);
    }
}

// ---------------------------------------------------------------------------
extern "C" void kernel_launch(void* const* d_in, const int* in_sizes, int n_in,
                              void* d_out, int out_size, void* d_ws, size_t ws_size,
                              hipStream_t stream) {
  const float* hs   = (const float*)d_in[0];
  const float* Wq   = (const float*)d_in[1];
  const float* Wk   = (const float*)d_in[2];
  const float* Wv   = (const float*)d_in[3];
  const float* Wo   = (const float*)d_in[4];
  const float* cosv = (const float*)d_in[5];
  const float* sinv = (const float*)d_in[6];
  const float* kst  = (const float*)d_in[7];
  const float* vst  = (const float*)d_in[8];
  const int*   lut  = (const int*)d_in[9];
  const int*   ptrs = (const int*)d_in[10];
  const int*   pfx  = (const int*)d_in[11];
  float*       out  = (float*)d_out;

  char* w = (char*)d_ws;
  size_t off = 0;
  auto carve = [&](size_t bytes) {
    void* p = w + off;
    off += (bytes + 255) & ~(size_t)255;
    return p;
  };
  const size_t M = (size_t)BSZ * QLEN;  // 2048
  bf16_t* q_lin  = (bf16_t*)carve(M * (NH * HD) * 2);
  bf16_t* k_lin  = (bf16_t*)carve(M * (NKV * HD) * 2);
  bf16_t* v_lin  = (bf16_t*)carve(M * (NKV * HD) * 2);
  bf16_t* q_rope = (bf16_t*)carve(M * (NH * HD) * 2);
  bf16_t* kctx   = (bf16_t*)carve((size_t)BSZ * NKV * SCTX * HD * 2);
  bf16_t* vctxT  = (bf16_t*)carve((size_t)BSZ * NKV * SCTX * HD * 2);
  bf16_t* attn   = (bf16_t*)carve(M * (NH * HD) * 2);
  (void)ws_size; (void)in_sizes; (void)n_in; (void)out_size;

  const dim3 blk(256);

  // QKV projections (A fp32, W fp32 staged by TDM -> bf16 out)
  gemm_kn<false, true><<<dim3((NH * HD) / 128, (int)M / 64), blk, 0, stream>>>(
      hs, Wq, q_lin, (int)M, NH * HD, HID);
  gemm_kn<false, true><<<dim3((NKV * HD) / 128, (int)M / 64), blk, 0, stream>>>(
      hs, Wk, k_lin, (int)M, NKV * HD, HID);
  gemm_kn<false, true><<<dim3((NKV * HD) / 128, (int)M / 64), blk, 0, stream>>>(
      hs, Wv, v_lin, (int)M, NKV * HD, HID);

  // RoPE(Q) into (b,h,s,d)
  rope_q_kernel<<<(unsigned)((M * NH * HD) / 256), blk, 0, stream>>>(
      q_lin, cosv, sinv, q_rope);

  // Paged KV gather + scatter of new blocks (K RoPE'd), V transposed
  build_kv_kernel<<<BSZ * NBLK * NKV, blk, 0, stream>>>(
      k_lin, v_lin, kst, vst, lut, ptrs, cosv, sinv, kctx, vctxT);

  // Flash attention
  attn_kernel<<<BSZ * NH * (QLEN / 64), blk, 0, stream>>>(
      q_rope, kctx, vctxT, pfx, attn);

  // O projection (A bf16, fp32 out)
  gemm_kn<true, false><<<dim3(HID / 128, (int)M / 64), blk, 0, stream>>>(
      attn, Wo, out, (int)M, HID, NH * HD);
}
